// MessagePassing_78889959293207
// MI455X (gfx1250) — matrix operand order repaired and checked by reading, not compiled, not run on hardware
//
#include <hip/hip_runtime.h>

// Problem constants (from reference)
#define Bc 4
#define Rc 10
#define Nc 80
#define Ec 20
#define Dc 300
#define Hc 512

// Padded K strides (multiples of 32 for WMMA k-tiles, of 16 for 32B-aligned B loads)
#define K1 640   // [adj | on] : 600 -> 640   (w1 cols permuted + zero padded)
#define K2 832   // [adj | a*edges] : 812 -> 832 (w2 zero padded)
#define KA 320   // adj : 300 -> 320           (wadj zero padded)

// Blocking: 4 nodes per block -> M = 80 rows = 5 exact 16-row WMMA tiles
#define NPB 4
#define ROWS (NPB * Ec)        // 80
#define NBLK (Nc / NPB)        // 20 node-blocks per (b,r)
#define MT 5                   // M-tiles per block
#define NTW 2                  // N-tiles per wave (16 waves x 2 x 16 = 512 cols)

typedef __attribute__((ext_vector_type(16))) __bf16 v16bf;
typedef __attribute__((ext_vector_type(8)))  float  v8f;

__device__ __forceinline__ unsigned short f32_to_bf16(float f) {
  unsigned int u = __float_as_uint(f);
  unsigned int r = u + 0x7FFFu + ((u >> 16) & 1u);   // round-to-nearest-even
  return (unsigned short)(r >> 16);
}
__device__ __forceinline__ float bf16_to_f32(unsigned short h) {
  return __uint_as_float(((unsigned int)h) << 16);
}
__device__ __forceinline__ unsigned int mul2_bf16(unsigned int packed, float q0, float q1) {
  float lo = __uint_as_float(packed << 16);
  float hi = __uint_as_float(packed & 0xFFFF0000u);
  lo *= q0; hi *= q1;
  return (unsigned int)f32_to_bf16(lo) | ((unsigned int)f32_to_bf16(hi) << 16);
}

// ---- A fragment (16x32 bf16) from row-major LDS  ----------------------------
// ISA 16-bit A layout: lanes 0-15 (M=lane) hold K {0..7, 16..23};
// lanes 16-31 (M=lane-16) hold K {8..15, 24..31}.
__device__ __forceinline__ v16bf load_a_frag(const unsigned short* base, int stride,
                                             int row0, int kBase, int lane) {
  int m  = row0 + (lane & 15);
  int hi = lane >> 4;
  const unsigned short* p = base + m * stride + kBase;
  union { uint4 q[2]; v16bf v; } u;
  u.q[0] = *(const uint4*)(p + hi * 8);
  u.q[1] = *(const uint4*)(p + 16 + hi * 8);
  return u.v;
}

// Same, with per-K scaling by q[k] fused in (for q * edges / q * u)
__device__ __forceinline__ v16bf load_a_frag_q(const unsigned short* base, int stride,
                                               int row0, int kBase, int lane,
                                               const float* qrow) {
  int m  = row0 + (lane & 15);
  int hi = lane >> 4;
  const unsigned short* p = base + m * stride + kBase;
  int c0 = kBase + hi * 8;
  int c1 = kBase + 16 + hi * 8;
  union { uint4 q[2]; unsigned int w[8]; v16bf v; } u;
  u.q[0] = *(const uint4*)(p + hi * 8);
  u.q[1] = *(const uint4*)(p + 16 + hi * 8);
  float4 qa = *(const float4*)(qrow + c0);
  float4 qb = *(const float4*)(qrow + c0 + 4);
  float4 qc = *(const float4*)(qrow + c1);
  float4 qd = *(const float4*)(qrow + c1 + 4);
  u.w[0] = mul2_bf16(u.w[0], qa.x, qa.y); u.w[1] = mul2_bf16(u.w[1], qa.z, qa.w);
  u.w[2] = mul2_bf16(u.w[2], qb.x, qb.y); u.w[3] = mul2_bf16(u.w[3], qb.z, qb.w);
  u.w[4] = mul2_bf16(u.w[4], qc.x, qc.y); u.w[5] = mul2_bf16(u.w[5], qc.z, qc.w);
  u.w[6] = mul2_bf16(u.w[6], qd.x, qd.y); u.w[7] = mul2_bf16(u.w[7], qd.z, qd.w);
  return u.v;
}

// ---- B fragment (32x16 bf16) from global weights W[out][K] (B = W^T) --------
__device__ __forceinline__ v16bf load_b_frag(const unsigned short* W, int strideK,
                                             int nBase, int kBase, int lane) {
  int n = nBase + (lane & 15);
  int k = kBase + ((lane >> 4) << 3);
  const unsigned short* p = W + (size_t)n * strideK + k;
  union { uint4 q[2]; v16bf v; } u;
  u.q[0] = *(const uint4*)(p);
  u.q[1] = *(const uint4*)(p + 16);
  return u.v;
}

// ---- GEMM core: per wave, C is 5 M-tiles x 2 N-tiles of 16x16 ---------------
template<bool QSCALE>
__device__ __forceinline__ void gemm_wave(const unsigned short* A, int strideA,
                                          const unsigned short* W, int strideW,
                                          int kTiles, int wave, int lane,
                                          const float* qrow, v8f C[MT][NTW]) {
  const int nW = wave * (NTW * 16);
  for (int kt = 0; kt < kTiles; ++kt) {
    const int kBase = kt * 32;
    v16bf a[MT];
#pragma unroll
    for (int i = 0; i < MT; ++i) {
      if constexpr (QSCALE)
        a[i] = load_a_frag_q(A, strideA, i * 16, kBase, lane, qrow);
      else
        a[i] = load_a_frag(A, strideA, i * 16, kBase, lane);
    }
#pragma unroll
    for (int j = 0; j < NTW; ++j) {
      v16bf b = load_b_frag(W, strideW, nW + j * 16, kBase, lane);
#pragma unroll
      for (int i = 0; i < MT; ++i)
        C[i][j] = __builtin_amdgcn_wmma_f32_16x16x32_bf16(false, a[i], false, b,
                                                          (short)0, C[i][j], false, false);
    }
  }
}

__device__ __forceinline__ void zero_acc(v8f C[MT][NTW]) {
#pragma unroll
  for (int i = 0; i < MT; ++i)
#pragma unroll
    for (int j = 0; j < NTW; ++j) C[i][j] = (v8f){};
}

// C layout: VGPR r, lanes 0-15 -> M=r, N=lane; lanes 16-31 -> M=r+8, N=lane-16
__device__ __forceinline__ void epilogue_bf16(v8f C[MT][NTW], unsigned short* dst,
                                              const float* bias, int wave, int lane) {
  const int nW = wave * (NTW * 16), lo = lane & 15, hiOff = (lane >> 4) * 8;
#pragma unroll
  for (int j = 0; j < NTW; ++j) {
    int n = nW + j * 16 + lo;
    float bs = bias[n];
#pragma unroll
    for (int i = 0; i < MT; ++i)
#pragma unroll
      for (int r = 0; r < 8; ++r) {
        int m = i * 16 + hiOff + r;
        dst[m * Hc + n] = f32_to_bf16(C[i][j][r] + bs);
      }
  }
}

// ============================ prep kernels ===================================
__global__ void convert_weights(const float* __restrict__ src, unsigned short* __restrict__ dst,
                                int rows, int srcK, int dstK, int perm) {
  int total = rows * dstK;
  for (int i = blockIdx.x * blockDim.x + threadIdx.x; i < total; i += gridDim.x * blockDim.x) {
    int o = i / dstK, k = i - o * dstK;
    float v = 0.f;
    if (k < srcK) {
      int ks = k;
      if (perm) ks = (k < perm) ? (k + perm) : (k - perm);   // swap [on|adj] -> [adj|on]
      v = src[(size_t)o * srcK + ks];
    }
    dst[i] = f32_to_bf16(v);
  }
}

__global__ void compute_q(const float* __restrict__ ques, const float* __restrict__ wqw,
                          const float* __restrict__ wqb, float* __restrict__ qf) {
  int br = blockIdx.x;
  for (int h = threadIdx.x; h < Hc; h += blockDim.x) {
    const float* x = ques + (size_t)br * Hc;
    const float* w = wqw + (size_t)h * Hc;
    float acc = wqb[h];
    for (int k = 0; k < Hc; ++k) acc += x[k] * w[k];
    qf[(size_t)br * Hc + h] = acc;
  }
}

// ============================ main kernel ====================================
// one block per (b, r, 4-node group); 16 waves; dynamic LDS:
//   sA u16[80][832]  (133120 B) : [adj | on] -> [adj | a*edges]
//   sE u16[80][512]  ( 81920 B) : edges, then u
//   sS u16[80][512]  ( 81920 B) : s / v scores (bf16), then softmax weights
// pBuf f32[80][512] (163840 B) aliases sA (+ start of sE) after they retire.
#define SMEM_BYTES (ROWS * K2 * 2 + ROWS * Hc * 2 + ROWS * Hc * 2)

__global__ void __launch_bounds__(512)
message_passing_kernel(const float* __restrict__ adj_list,
                       const float* __restrict__ original_nodes,
                       const unsigned short* __restrict__ w1p,  const float* __restrict__ w1b,
                       const unsigned short* __restrict__ weW,  const float* __restrict__ web,
                       const unsigned short* __restrict__ w2W,  const float* __restrict__ w2b,
                       const unsigned short* __restrict__ wvW,  const float* __restrict__ wvb,
                       const unsigned short* __restrict__ wadjW,const float* __restrict__ wadjb,
                       const float* __restrict__ qf,
                       float* __restrict__ out) {
  extern __shared__ char smem[];
  unsigned short* sA = (unsigned short*)smem;                 // [ROWS][K2]
  unsigned short* sE = sA + ROWS * K2;                        // [ROWS][Hc]
  unsigned short* sS = sE + ROWS * Hc;                        // [ROWS][Hc]
  float*          pBuf = (float*)smem;                        // [ROWS][Hc] (aliases sA/sE)

  const int tid  = threadIdx.x;
  const int lane = tid & 31;
  const int wave = tid >> 5;

  const int bid = blockIdx.x;
  const int bI  = bid / (Rc * NBLK);
  const int rem = bid - bI * (Rc * NBLK);
  const int rI  = rem / NBLK;
  const int nb  = rem - rI * NBLK;

  // rows (node_in_block*20 + e) are contiguous in adj_list memory
  const float* adjB = adj_list + (((size_t)bI * Rc + rI) * Nc + nb * NPB) * Ec * Dc;
  const float* onB  = original_nodes + (((size_t)bI * Rc + rI) * Nc + nb * NPB) * Dc;
  const float* qrow = qf + ((size_t)bI * Rc + rI) * Hc;
  float*       outB = out + (((size_t)bI * Rc + rI) * Nc + nb * NPB) * Hc;

  // ---- zero sA (kills NaN risk in zero-padded K columns), then stage inputs
  {
    unsigned int* z = (unsigned int*)sA;
    for (int i = tid; i < ROWS * K2 / 2; i += 512) z[i] = 0u;
  }
  __syncthreads();
  for (int i = tid; i < ROWS * Dc; i += 512) {
    int row = i / Dc, k = i - row * Dc;
    int g = row / Ec;                                  // node within block
    sA[row * K2 + k]      = f32_to_bf16(adjB[(size_t)row * Dc + k]); // adj cols [0,300)
    sA[row * K2 + Dc + k] = f32_to_bf16(onB[(size_t)g * Dc + k]);    // on  cols [300,600)
  }
  __syncthreads();

  v8f C[MT][NTW];

  // ---- GEMM1: edges = [adj|on] @ w1p^T + b1  -> sE (bf16)
  zero_acc(C);
  gemm_wave<false>(sA, K2, w1p, K1, K1 / 32, wave, lane, qrow, C);
  epilogue_bf16(C, sE, w1b, wave, lane);
  __syncthreads();

  // ---- GEMM2: s = (q*edges) @ we^T + be  -> sS (bf16 scores)
  zero_acc(C);
  gemm_wave<true>(sE, Hc, weW, Hc, Hc / 32, wave, lane, qrow, C);
  epilogue_bf16(C, sS, web, wave, lane);
  __syncthreads();

  // ---- softmax over E per (node, column); a*edges -> sA cols [300, 812)
  for (int p = tid; p < NPB * Hc; p += 512) {
    int g = p >> 9, c = p & (Hc - 1);
    int rb = g * Ec;
    float mx = -3.4e38f;
#pragma unroll
    for (int e = 0; e < Ec; ++e) mx = fmaxf(mx, bf16_to_f32(sS[(rb + e) * Hc + c]));
    float ex[Ec]; float sum = 0.f;
#pragma unroll
    for (int e = 0; e < Ec; ++e) {
      ex[e] = __expf(bf16_to_f32(sS[(rb + e) * Hc + c]) - mx);
      sum += ex[e];
    }
    float inv = __frcp_rn(sum);
#pragma unroll
    for (int e = 0; e < Ec; ++e) {
      float a = ex[e] * inv;
      sA[(rb + e) * K2 + Dc + c] = f32_to_bf16(a * bf16_to_f32(sE[(rb + e) * Hc + c]));
    }
  }
  __syncthreads();

  // ---- GEMM-w2: u = [adj | a*edges] @ w2^T + b2 -> sE (bf16)
  zero_acc(C);
  gemm_wave<false>(sA, K2, w2W, K2, K2 / 32, wave, lane, qrow, C);
  epilogue_bf16(C, sE, w2b, wave, lane);
  __syncthreads();

  // ---- GEMM-wv: v = (q*u) @ wv^T + bv -> sS (bf16 scores)
  zero_acc(C);
  gemm_wave<true>(sE, Hc, wvW, Hc, Hc / 32, wave, lane, qrow, C);
  epilogue_bf16(C, sS, wvb, wave, lane);
  __syncthreads();

  // ---- softmax over E per (node, column) -> b weights (bf16) in sS
  for (int p = tid; p < NPB * Hc; p += 512) {
    int g = p >> 9, c = p & (Hc - 1);
    int rb = g * Ec;
    float mx = -3.4e38f;
#pragma unroll
    for (int e = 0; e < Ec; ++e) mx = fmaxf(mx, bf16_to_f32(sS[(rb + e) * Hc + c]));
    float ex[Ec]; float sum = 0.f;
#pragma unroll
    for (int e = 0; e < Ec; ++e) {
      ex[e] = __expf(bf16_to_f32(sS[(rb + e) * Hc + c]) - mx);
      sum += ex[e];
    }
    float inv = __frcp_rn(sum);
#pragma unroll
    for (int e = 0; e < Ec; ++e) sS[(rb + e) * Hc + c] = f32_to_bf16(ex[e] * inv);
  }
  __syncthreads();

  // ---- GEMM-wadj: adjp = adj @ wadj^T (K=320, wadj zero-padded over a*edges cols 300..319)
  zero_acc(C);
  gemm_wave<false>(sA, K2, wadjW, KA, KA / 32, wave, lane, qrow, C);
  __syncthreads();   // all A reads of sA done before pBuf overwrites it

  // write b * adjp products (f32) into pBuf (aliases retired sA/sE)
  {
    const int nW = wave * (NTW * 16), lo = lane & 15, hiOff = (lane >> 4) * 8;
#pragma unroll
    for (int j = 0; j < NTW; ++j) {
      int n = nW + j * 16 + lo;
#pragma unroll
      for (int i = 0; i < MT; ++i)
#pragma unroll
        for (int r = 0; r < 8; ++r) {
          int m = i * 16 + hiOff + r;
          pBuf[m * Hc + n] = C[i][j][r] * bf16_to_f32(sS[m * Hc + n]);
        }
    }
  }
  __syncthreads();

  // ---- reduce over E per node; softmax sums to 1 so +wadj_b folds in directly
  for (int p = tid; p < NPB * Hc; p += 512) {
    int g = p >> 9, c = p & (Hc - 1);
    int rb = g * Ec;
    float s = 0.f;
#pragma unroll
    for (int e = 0; e < Ec; ++e) s += pBuf[(rb + e) * Hc + c];
    outB[(size_t)g * Hc + c] = s + wadjb[c];
  }
}

// ============================ host launcher ==================================
// workspace layout (bytes, all 256-aligned)
#define OFF_W1   0u
#define OFF_WE   655360u     // 512*640*2
#define OFF_W2   1179648u    // + 512*512*2
#define OFF_WV   2031616u    // + 512*832*2
#define OFF_WADJ 2555904u    // + 512*512*2
#define OFF_Q    2883584u    // + 512*320*2   (q: 40*512*4 = 81920 -> end 2965504)

extern "C" void kernel_launch(void* const* d_in, const int* in_sizes, int n_in,
                              void* d_out, int out_size, void* d_ws, size_t ws_size,
                              hipStream_t stream) {
  const float* ques  = (const float*)d_in[0];
  const float* adj   = (const float*)d_in[1];
  const float* onod  = (const float*)d_in[2];
  const float* w1w   = (const float*)d_in[3];
  const float* w1b   = (const float*)d_in[4];
  const float* wqw   = (const float*)d_in[5];
  const float* wqb   = (const float*)d_in[6];
  const float* wew   = (const float*)d_in[7];
  const float* web   = (const float*)d_in[8];
  const float* w2w   = (const float*)d_in[9];
  const float* w2b   = (const float*)d_in[10];
  const float* wvw   = (const float*)d_in[11];
  const float* wvb   = (const float*)d_in[12];
  const float* wadjw = (const float*)d_in[13];
  const float* wadjb = (const float*)d_in[14];
  float* out = (float*)d_out;

  char* ws = (char*)d_ws;
  unsigned short* w1p_bf  = (unsigned short*)(ws + OFF_W1);
  unsigned short* we_bf   = (unsigned short*)(ws + OFF_WE);
  unsigned short* w2_bf   = (unsigned short*)(ws + OFF_W2);
  unsigned short* wv_bf   = (unsigned short*)(ws + OFF_WV);
  unsigned short* wadj_bf = (unsigned short*)(ws + OFF_WADJ);
  float*          q_f32   = (float*)(ws + OFF_Q);

  // weight conversion to zero-padded bf16 (L2-resident afterwards)
  convert_weights<<<(Hc * K1 + 255) / 256, 256, 0, stream>>>(w1w,  w1p_bf,  Hc, 2 * Dc,  K1, Dc);
  convert_weights<<<(Hc * Hc + 255) / 256, 256, 0, stream>>>(wew,  we_bf,   Hc, Hc,      Hc, 0);
  convert_weights<<<(Hc * K2 + 255) / 256, 256, 0, stream>>>(w2w,  w2_bf,   Hc, Dc + Hc, K2, 0);
  convert_weights<<<(Hc * Hc + 255) / 256, 256, 0, stream>>>(wvw,  wv_bf,   Hc, Hc,      Hc, 0);
  convert_weights<<<(Hc * KA + 255) / 256, 256, 0, stream>>>(wadjw, wadj_bf, Hc, Dc,     KA, 0);
  compute_q<<<Bc * Rc, 256, 0, stream>>>(ques, wqw, wqb, q_f32);

  (void)hipFuncSetAttribute((const void*)message_passing_kernel,
                            hipFuncAttributeMaxDynamicSharedMemorySize, SMEM_BYTES);

  message_passing_kernel<<<Bc * Rc * NBLK, 512, SMEM_BYTES, stream>>>(
      adj, onod,
      w1p_bf, w1b, we_bf, web, w2_bf, w2b, wv_bf, wvb, wadj_bf, wadjb,
      q_f32, out);

  (void)in_sizes; (void)n_in; (void)out_size; (void)ws_size;
}
